// CapECG_77498389889579
// MI455X (gfx1250) — compile-verified, matrix operand order checked by reading, not run on hardware
//
#include <hip/hip_runtime.h>

#define BATCH 64
#define SEQL  2148
#define DIN   10
#define LEN1  714
#define LCAPS 355
#define NROUTE 11360

typedef __attribute__((ext_vector_type(16))) _Float16 v16h;
typedef __attribute__((ext_vector_type(8)))  float    v8f;

// ---------------------------------------------------------------------------
// WMMA fragment load from an LDS row holding 32 contiguous K-halfs.
// 16-bit A (16x32) layout (wave32): lane l -> row = l&15, khalf = l>>4;
// element e: k = e + 8*khalf        for e in [0,8)
//            k = e + 8 + 8*khalf    for e in [8,16)
// -> two contiguous 8-half (16B) chunks; B is staged transposed so it uses
// the same loader.
// ---------------------------------------------------------------------------
__device__ __forceinline__ v16h frag_from_row(const _Float16* row, int kh) {
  v16h f;
  const _Float16* p = row + kh * 8;
#pragma unroll
  for (int i = 0; i < 8; ++i) { f[i] = p[i]; f[i + 8] = p[16 + i]; }
  return f;
}

// ---------------------------------------------------------------------------
// Elementwise helpers
// ---------------------------------------------------------------------------
__global__ void f32_to_f16_kernel(const float* __restrict__ src,
                                  _Float16* __restrict__ dst, int n) {
  int i = blockIdx.x * 256 + threadIdx.x;
  if (i < n) dst[i] = (_Float16)src[i];
}

__global__ void zero_f32_kernel(float* p, int n) {
  int i = blockIdx.x * 256 + threadIdx.x;
  if (i < n) p[i] = 0.f;
}

// ---------------------------------------------------------------------------
// Embedding: h = x @ emb_w.T + emb_b ; per-row max / mean over 32 channels.
// One wave per (b,l) row.
// ---------------------------------------------------------------------------
__global__ void embed_kernel(const float* __restrict__ x,
                             const float* __restrict__ W,   // (32,10)
                             const float* __restrict__ bias,
                             float* __restrict__ h,         // (B*L,32)
                             float* __restrict__ pooled) {  // (B,2,L)
  int lane = threadIdx.x & 31;
  int warp = threadIdx.x >> 5;
  size_t row = (size_t)blockIdx.x * 8 + warp;
  if (row >= (size_t)BATCH * SEQL) return;
  const float* xr = x + row * DIN;
  float acc = bias[lane];
#pragma unroll
  for (int d = 0; d < DIN; ++d) acc += xr[d] * W[lane * DIN + d];
  h[row * 32 + lane] = acc;
  float mx = acc, sm = acc;
#pragma unroll
  for (int o = 16; o > 0; o >>= 1) {
    mx = fmaxf(mx, __shfl_xor(mx, o, 32));
    sm += __shfl_xor(sm, o, 32);
  }
  if (lane == 0) {
    int b = (int)(row / SEQL), l = (int)(row % SEQL);
    pooled[((size_t)b * 2 + 0) * SEQL + l] = mx;
    pooled[((size_t)b * 2 + 1) * SEQL + l] = sm * (1.f / 32.f);
  }
}

// ---------------------------------------------------------------------------
// Attention conv (k=7,pad=3) + sigmoid; writes x_att output (tiled) and
// h_att in (B,32,L) f16 layout for the conv1 implicit GEMM.
// ---------------------------------------------------------------------------
__global__ void att_kernel(const float* __restrict__ h,
                           const float* __restrict__ pooled,
                           const float* __restrict__ w,   // (1,2,7) flat 14
                           const float* __restrict__ cb,  // (1,)
                           float* __restrict__ xatt,      // (B,L,32)
                           _Float16* __restrict__ hatt) { // (B,32,L)
  int i = blockIdx.x * 256 + threadIdx.x;
  if (i >= BATCH * SEQL) return;
  int b = i / SEQL, l = i % SEQL;
  float acc = cb[0];
#pragma unroll
  for (int c = 0; c < 2; ++c)
#pragma unroll
    for (int kk = 0; kk < 7; ++kk) {
      int ll = l + kk - 3;
      if (ll >= 0 && ll < SEQL)
        acc += pooled[((size_t)b * 2 + c) * SEQL + ll] * w[c * 7 + kk];
    }
  float att = 1.f / (1.f + __expf(-acc));
  const float* hr = h + (size_t)i * 32;
  float* xo = xatt + (size_t)i * 32;
#pragma unroll
  for (int c = 0; c < 32; ++c) {
    xo[c] = att;
    hatt[((size_t)b * 32 + c) * SEQL + l] = (_Float16)(hr[c] * att);
  }
}

// ---------------------------------------------------------------------------
// conv1: (B,32,2148) -> (B,32,714), k=9 stride=3, as implicit GEMM
// M=32 (outC), N=714, K=288.  One wave per 16x16 tile.
// ---------------------------------------------------------------------------
__global__ void conv1_wmma_kernel(const _Float16* __restrict__ W,  // (32,288)
                                  const _Float16* __restrict__ X,  // (B,32,2148)
                                  const float* __restrict__ bias,
                                  _Float16* __restrict__ Y) {      // (B,32,714)
  __shared__ _Float16 lA[512];
  __shared__ _Float16 lB[512];
  int lane = threadIdx.x;
  int n0 = blockIdx.x * 16, m0 = blockIdx.y * 16, b = blockIdx.z;
  v8f acc = {};
  for (int k0 = 0; k0 < 288; k0 += 32) {
    __syncthreads();
#pragma unroll
    for (int e = 0; e < 16; ++e) {
      int idx = lane * 16 + e;
      int r = idx >> 5, kloc = idx & 31;
      int kg = k0 + kloc;
      lA[idx] = W[(m0 + r) * 288 + kg];
      int t = n0 + r;
      int ci = kg / 9, kk = kg - ci * 9;
      lB[idx] = (t < LEN1)
                    ? X[((size_t)b * 32 + ci) * SEQL + t * 3 + kk]
                    : (_Float16)0.f;
    }
    __syncthreads();
    v16h a = frag_from_row(&lA[(lane & 15) * 32], lane >> 4);
    v16h bf = frag_from_row(&lB[(lane & 15) * 32], lane >> 4);
    acc = __builtin_amdgcn_wmma_f32_16x16x32_f16(false, a, false, bf,
                                                 (short)0, acc, false, false);
  }
  int t = n0 + (lane & 15);
  int mb = m0 + ((lane >> 4) * 8);
  if (t < LEN1) {
#pragma unroll
    for (int r = 0; r < 8; ++r)
      Y[((size_t)b * 32 + mb + r) * LEN1 + t] = (_Float16)(acc[r] + bias[mb + r]);
  }
}

// ---------------------------------------------------------------------------
// Caps conv: (B,32,714) -> (B,256,355), k=5 stride=2, implicit GEMM with the
// caps squash fused into the epilogue (each lane holds all 8 dims of one
// route in its accumulator VGPRs).  Writes prim (B,11360,8) f32.
// ---------------------------------------------------------------------------
__global__ void caps_wmma_kernel(const _Float16* __restrict__ W,  // (256,160)
                                 const _Float16* __restrict__ X,  // (B,32,714)
                                 const float* __restrict__ bias,  // (256,)
                                 float* __restrict__ prim) {
  __shared__ _Float16 lA[512];
  __shared__ _Float16 lB[512];
  int lane = threadIdx.x;
  int n0 = blockIdx.x * 16, m0 = blockIdx.y * 16, b = blockIdx.z;
  v8f acc = {};
  for (int k0 = 0; k0 < 160; k0 += 32) {
    __syncthreads();
#pragma unroll
    for (int e = 0; e < 16; ++e) {
      int idx = lane * 16 + e;
      int r = idx >> 5, kloc = idx & 31;
      int kg = k0 + kloc;
      lA[idx] = W[(m0 + r) * 160 + kg];
      int t = n0 + r;
      int ci = kg / 5, kk = kg - ci * 5;
      lB[idx] = (t < LCAPS)
                    ? X[((size_t)b * 32 + ci) * LEN1 + t * 2 + kk]
                    : (_Float16)0.f;
    }
    __syncthreads();
    v16h a = frag_from_row(&lA[(lane & 15) * 32], lane >> 4);
    v16h bf = frag_from_row(&lB[(lane & 15) * 32], lane >> 4);
    acc = __builtin_amdgcn_wmma_f32_16x16x32_f16(false, a, false, bf,
                                                 (short)0, acc, false, false);
  }
  int t = n0 + (lane & 15);
  int mb = m0 + ((lane >> 4) * 8);  // multiple of 8 -> one capsule per lane
  if (t < LCAPS) {
    float v[8];
    float n2 = 0.f;
#pragma unroll
    for (int r = 0; r < 8; ++r) { v[r] = acc[r] + bias[mb + r]; n2 += v[r] * v[r]; }
    float f = n2 / (1.f + n2) * rsqrtf(n2);
    int cap = mb >> 3;
    float* P = prim + (((size_t)b * NROUTE + (size_t)cap * LCAPS + t) * 8);
#pragma unroll
    for (int r = 0; r < 8; ++r) P[r] = f * v[r];
  }
}

// ---------------------------------------------------------------------------
// Routing pass 1: usum[b,k,z] = sum_j sum_i prim[b,j,i]*rW[k,j,i,z]
// Block = 256 routes of one batch; LDS reduction then one atomic per value.
// ---------------------------------------------------------------------------
__global__ void routing_pass1(const float* __restrict__ prim,
                              const float* __restrict__ rW,
                              float* __restrict__ usum) {
  int b = blockIdx.y;
  int j = blockIdx.x * 256 + threadIdx.x;
  __shared__ float sacc[32];
  if (threadIdx.x < 32) sacc[threadIdx.x] = 0.f;
  __syncthreads();
  if (j < NROUTE) {
    float p[8];
#pragma unroll
    for (int i = 0; i < 8; ++i) p[i] = prim[((size_t)b * NROUTE + j) * 8 + i];
    float acc[32];
#pragma unroll
    for (int t = 0; t < 32; ++t) acc[t] = 0.f;
#pragma unroll
    for (int k = 0; k < 2; ++k)
#pragma unroll
      for (int i = 0; i < 8; ++i) {
        const float* w = rW + (((size_t)k * NROUTE + j) * 8 + i) * 16;
        float pv = p[i];
#pragma unroll
        for (int z = 0; z < 16; ++z) acc[k * 16 + z] += pv * w[z];
      }
#pragma unroll
    for (int t = 0; t < 32; ++t) atomicAdd(&sacc[t], acc[t]);
  }
  __syncthreads();
  if (threadIdx.x < 32) atomicAdd(&usum[b * 32 + threadIdx.x], sacc[threadIdx.x]);
}

// ---------------------------------------------------------------------------
// Routing pass 2: c = softmax_k(<u,usum>/4) + rb ;  s[b,k,z] += u*c
// ---------------------------------------------------------------------------
__global__ void routing_pass2(const float* __restrict__ prim,
                              const float* __restrict__ rW,
                              const float* __restrict__ usum,
                              const float* __restrict__ rb,
                              float* __restrict__ sbuf) {
  int b = blockIdx.y;
  int j = blockIdx.x * 256 + threadIdx.x;
  __shared__ float sacc[32];
  if (threadIdx.x < 32) sacc[threadIdx.x] = 0.f;
  __syncthreads();
  if (j < NROUTE) {
    float p[8];
#pragma unroll
    for (int i = 0; i < 8; ++i) p[i] = prim[((size_t)b * NROUTE + j) * 8 + i];
    float u[32];
#pragma unroll
    for (int t = 0; t < 32; ++t) u[t] = 0.f;
#pragma unroll
    for (int k = 0; k < 2; ++k)
#pragma unroll
      for (int i = 0; i < 8; ++i) {
        const float* w = rW + (((size_t)k * NROUTE + j) * 8 + i) * 16;
        float pv = p[i];
#pragma unroll
        for (int z = 0; z < 16; ++z) u[k * 16 + z] += pv * w[z];
      }
    float d0 = 0.f, d1 = 0.f;
#pragma unroll
    for (int z = 0; z < 16; ++z) {
      d0 += u[z] * usum[b * 32 + z];
      d1 += u[16 + z] * usum[b * 32 + 16 + z];
    }
    d0 *= 0.25f; d1 *= 0.25f;
    float m = fmaxf(d0, d1);
    float e0 = __expf(d0 - m), e1 = __expf(d1 - m);
    float inv = 1.f / (e0 + e1);
    float c0 = e0 * inv + rb[j];
    float c1 = e1 * inv + rb[NROUTE + j];
    float acc[32];
#pragma unroll
    for (int z = 0; z < 16; ++z) { acc[z] = u[z] * c0; acc[16 + z] = u[16 + z] * c1; }
#pragma unroll
    for (int t = 0; t < 32; ++t) atomicAdd(&sacc[t], acc[t]);
  }
  __syncthreads();
  if (threadIdx.x < 32) atomicAdd(&sbuf[b * 32 + threadIdx.x], sacc[threadIdx.x]);
}

// ---------------------------------------------------------------------------
// squash_exp(s) -> x1 (f32 + f16)
// ---------------------------------------------------------------------------
__global__ void routing_final(const float* __restrict__ s,
                              float* __restrict__ x1f,
                              _Float16* __restrict__ x1h) {
  int b = blockIdx.x, t = threadIdx.x;  // 32 threads
  float v = s[b * 32 + t];
  float sq = v * v;
#pragma unroll
  for (int o = 1; o < 16; o <<= 1) sq += __shfl_xor(sq, o, 32);  // within k-group
  float n = sqrtf(sq);
  float out = (1.f - 1.f / (__expf(n) + 1e-20f)) * v / (n + 1e-20f);
  x1f[b * 32 + t] = out;
  x1h[b * 32 + t] = (_Float16)out;
}

// ---------------------------------------------------------------------------
// Generic WMMA GEMM: C[M,N] = act(A[M,K] * W[N,K]^T + bias[N])
// One wave per 16x16 tile; A/W tiles staged through LDS.
// ---------------------------------------------------------------------------
__global__ void gemm_wmma_kernel(const _Float16* __restrict__ A, int lda,
                                 const _Float16* __restrict__ W, int ldw,
                                 const float* __restrict__ bias,
                                 float* __restrict__ Cf, _Float16* __restrict__ Ch,
                                 int M, int N, int K, int act) {
  __shared__ _Float16 lA[512];
  __shared__ _Float16 lB[512];
  int lane = threadIdx.x;
  int n0 = blockIdx.x * 16, m0 = blockIdx.y * 16;
  v8f acc = {};
  for (int k0 = 0; k0 < K; k0 += 32) {
    if (k0 + 32 < K) {
      int pn = n0 + (lane & 15); if (pn > N - 1) pn = N - 1;
      __builtin_prefetch(&W[(size_t)pn * ldw + k0 + 32], 0, 1);
    }
    __syncthreads();
#pragma unroll
    for (int e = 0; e < 16; ++e) {
      int idx = lane * 16 + e;
      int r = idx >> 5, kloc = idx & 31;
      int kg = k0 + kloc;
      int ma = m0 + r; if (ma > M - 1) ma = M - 1;
      int nb = n0 + r; if (nb > N - 1) nb = N - 1;
      lA[idx] = (kg < K) ? A[(size_t)ma * lda + kg] : (_Float16)0.f;
      lB[idx] = (kg < K) ? W[(size_t)nb * ldw + kg] : (_Float16)0.f;
    }
    __syncthreads();
    v16h a = frag_from_row(&lA[(lane & 15) * 32], lane >> 4);
    v16h bf = frag_from_row(&lB[(lane & 15) * 32], lane >> 4);
    acc = __builtin_amdgcn_wmma_f32_16x16x32_f16(false, a, false, bf,
                                                 (short)0, acc, false, false);
  }
  int n = n0 + (lane & 15);
  int mb = m0 + ((lane >> 4) * 8);
  if (n < N) {
#pragma unroll
    for (int r = 0; r < 8; ++r) {
      int m = mb + r;
      if (m < M) {
        float v = acc[r] + (bias ? bias[n] : 0.f);
        if (act == 1) v = fmaxf(v, 0.f);
        else if (act == 2) v = 1.f / (1.f + __expf(-v));
        if (Cf) Cf[(size_t)m * N + n] = v;
        if (Ch) Ch[(size_t)m * N + n] = (_Float16)v;
      }
    }
  }
}

// ---------------------------------------------------------------------------
// DenseNet conv (k=1 or 3, pad), implicit GEMM with fused BN(eval)+ReLU on
// the input during LDS staging.  out raw (concat semantics), f32.
// ---------------------------------------------------------------------------
__global__ void dense_conv_wmma_kernel(const _Float16* __restrict__ W,  // (Mo,K)
                                       const float* __restrict__ feat,  // (B,CtotIn,L)
                                       const float* __restrict__ bnG,
                                       const float* __restrict__ bnB,
                                       const float* __restrict__ bias,  // or null
                                       float* __restrict__ out,         // (B,CtotOut,L)
                                       int Cin, int CtotIn, int CtotOut, int Coff,
                                       int Mo, int L, int ks, int pad, int K) {
  __shared__ _Float16 lA[512];
  __shared__ _Float16 lB[512];
  const float kInvStd = 0.99999500003749968f;  // 1/sqrt(1+1e-5)
  int lane = threadIdx.x;
  int n0 = blockIdx.x * 16, m0 = blockIdx.y * 16, b = blockIdx.z;
  v8f acc = {};
  for (int k0 = 0; k0 < K; k0 += 32) {
    __syncthreads();
#pragma unroll
    for (int e = 0; e < 16; ++e) {
      int idx = lane * 16 + e;
      int r = idx >> 5, kloc = idx & 31;
      int kg = k0 + kloc;
      int ma = m0 + r; if (ma > Mo - 1) ma = Mo - 1;
      lA[idx] = (kg < K) ? W[(size_t)ma * K + kg] : (_Float16)0.f;
      float bv = 0.f;
      if (kg < K) {
        int ci = kg / ks, kk = kg - ci * ks;
        int l = n0 + r + kk - pad;
        if (l >= 0 && l < L) {
          float xv = feat[((size_t)b * CtotIn + ci) * L + l];
          xv = xv * (bnG[ci] * kInvStd) + bnB[ci];
          bv = xv > 0.f ? xv : 0.f;
        }
      }
      lB[idx] = (_Float16)bv;
    }
    __syncthreads();
    v16h a = frag_from_row(&lA[(lane & 15) * 32], lane >> 4);
    v16h bf = frag_from_row(&lB[(lane & 15) * 32], lane >> 4);
    acc = __builtin_amdgcn_wmma_f32_16x16x32_f16(false, a, false, bf,
                                                 (short)0, acc, false, false);
  }
  int t = n0 + (lane & 15);
  int mb = m0 + ((lane >> 4) * 8);
  if (t < L) {
#pragma unroll
    for (int r = 0; r < 8; ++r) {
      int m = mb + r;
      if (m < Mo)
        out[((size_t)b * CtotOut + Coff + m) * L + t] =
            acc[r] + (bias ? bias[m] : 0.f);
    }
  }
}

// ---------------------------------------------------------------------------
// avg_pool1d(k=9, s=5)
// ---------------------------------------------------------------------------
__global__ void avgpool_kernel(const float* __restrict__ in, float* __restrict__ out,
                               int C, int L, int Lo, int CtotOut) {
  int i = blockIdx.x * 256 + threadIdx.x;
  int total = BATCH * C * Lo;
  if (i >= total) return;
  int lo = i % Lo;
  int c = (i / Lo) % C;
  int b = i / (Lo * C);
  const float* p = in + ((size_t)b * C + c) * L + lo * 5;
  float s = 0.f;
#pragma unroll
  for (int j = 0; j < 9; ++j) s += p[j];
  out[((size_t)b * CtotOut + c) * Lo + lo] = s * (1.f / 9.f);
}

// ---------------------------------------------------------------------------
// x_gen (B,L,D) -> feat (B,10,L) with CtotOut=138 stride (block-1 layout)
// ---------------------------------------------------------------------------
__global__ void feat_init_kernel(const float* __restrict__ xgen,
                                 float* __restrict__ feat) {
  int i = blockIdx.x * 256 + threadIdx.x;
  int total = BATCH * DIN * SEQL;
  if (i >= total) return;
  int l = i % SEQL;
  int d = (i / SEQL) % DIN;
  int b = i / (SEQL * DIN);
  feat[((size_t)b * 138 + d) * SEQL + l] = xgen[((size_t)b * SEQL + l) * DIN + d];
}

// ---------------------------------------------------------------------------
// out1 head: relu(x1 @ Wa.T + ba) @ Wb.T + bb
// ---------------------------------------------------------------------------
__global__ void head1_kernel(const float* __restrict__ x1,
                             const float* __restrict__ Wa, const float* __restrict__ ba,
                             const float* __restrict__ Wb, const float* __restrict__ bb,
                             float* __restrict__ out) {
  int b = blockIdx.x, t = threadIdx.x;  // 32 threads
  float acc = ba[t];
#pragma unroll
  for (int k = 0; k < 32; ++k) acc += x1[b * 32 + k] * Wa[t * 32 + k];
  acc = fmaxf(acc, 0.f) * Wb[t];
#pragma unroll
  for (int o = 16; o > 0; o >>= 1) acc += __shfl_xor(acc, o, 32);
  if (t == 0) out[b] = acc + bb[0];
}

// ---------------------------------------------------------------------------
// x2 head: relu(flat848 @ Wa.T + ba) @ Wb.T + bb
// ---------------------------------------------------------------------------
__global__ void head2_kernel(const float* __restrict__ feat,  // (B,53,16)
                             const float* __restrict__ Wa, const float* __restrict__ ba,
                             const float* __restrict__ Wb, const float* __restrict__ bb,
                             float* __restrict__ out) {
  int b = blockIdx.x, t = threadIdx.x;  // 64 threads
  __shared__ float red[64];
  const float* f = feat + (size_t)b * 848;
  float acc = ba[t];
  for (int k = 0; k < 848; ++k) acc += f[k] * Wa[t * 848 + k];
  red[t] = fmaxf(acc, 0.f) * Wb[t];
  __syncthreads();
  if (t == 0) {
    float s = 0.f;
    for (int i = 0; i < 64; ++i) s += red[i];
    out[b] = s + bb[0];
  }
}

// ---------------------------------------------------------------------------
// Host side
// ---------------------------------------------------------------------------
extern "C" void kernel_launch(void* const* d_in, const int* in_sizes, int n_in,
                              void* d_out, int out_size, void* d_ws, size_t ws_size,
                              hipStream_t stream) {
  (void)in_sizes; (void)n_in; (void)out_size; (void)ws_size;
  const float* x       = (const float*)d_in[0];
  const float* emb_w   = (const float*)d_in[1];
  const float* emb_b   = (const float*)d_in[2];
  const float* conv2_w = (const float*)d_in[3];
  const float* conv2_b = (const float*)d_in[4];
  const float* conv1_w = (const float*)d_in[5];
  const float* conv1_b = (const float*)d_in[6];
  const float* caps_w  = (const float*)d_in[7];
  const float* caps_b  = (const float*)d_in[8];
  const float* rW      = (const float*)d_in[9];
  const float* rb      = (const float*)d_in[10];
  const float* fc1_w   = (const float*)d_in[11];
  const float* fc1_b   = (const float*)d_in[12];
  const float* fc2_w   = (const float*)d_in[13];
  const float* fc2_b   = (const float*)d_in[14];
  const float* fc3_w   = (const float*)d_in[15];
  const float* fc3_b   = (const float*)d_in[16];
  const float* lin1a_w = (const float*)d_in[17];
  const float* lin1a_b = (const float*)d_in[18];
  const float* lin1b_w = (const float*)d_in[19];
  const float* lin1b_b = (const float*)d_in[20];
  const int denseBase = 21;
  const float* lin2a_w = (const float*)d_in[78];
  const float* lin2a_b = (const float*)d_in[79];
  const float* lin2b_w = (const float*)d_in[80];
  const float* lin2b_b = (const float*)d_in[81];

  float* out   = (float*)d_out;
  float* out1  = out;                                   // (B,1)
  float* xatt  = out + 64;                              // (B,L,32)
  float* xgen  = xatt + (size_t)BATCH * SEQL * 32;      // (B,L,D)
  float* x2    = xgen + (size_t)BATCH * SEQL * DIN;     // (B,1)

  char* ws = (char*)d_ws;
  size_t off = 0;
  auto alloc = [&](size_t bytes) -> void* {
    off = (off + 255) & ~(size_t)255;
    void* p = ws + off;
    off += bytes;
    return p;
  };

  float*    h      = (float*)alloc((size_t)BATCH * SEQL * 32 * 4);
  float*    pooled = (float*)alloc((size_t)BATCH * 2 * SEQL * 4);
  _Float16* hatt   = (_Float16*)alloc((size_t)BATCH * 32 * SEQL * 2);
  _Float16* h1     = (_Float16*)alloc((size_t)BATCH * 32 * LEN1 * 2);
  float*    prim   = (float*)alloc((size_t)BATCH * NROUTE * 8 * 4);
  float*    usum   = (float*)alloc(64 * 32 * 4);
  float*    sbuf   = (float*)alloc(64 * 32 * 4);
  float*    x1f    = (float*)alloc(64 * 32 * 4);
  _Float16* x1h    = (_Float16*)alloc(64 * 32 * 2);
  float*    g1f    = (float*)alloc((size_t)64 * 512 * 4);
  _Float16* g1h    = (_Float16*)alloc((size_t)64 * 512 * 2);
  float*    g2f    = (float*)alloc((size_t)64 * 1024 * 4);
  _Float16* g2h    = (_Float16*)alloc((size_t)64 * 1024 * 2);
  float*    featA  = (float*)alloc((size_t)BATCH * 138 * SEQL * 4);
  float*    featT  = (float*)alloc((size_t)BATCH * 41 * SEQL * 4);

  auto convert = [&](const float* src, size_t n) -> _Float16* {
    _Float16* dst = (_Float16*)alloc(n * 2);
    int blocks = (int)((n + 255) / 256);
    f32_to_f16_kernel<<<blocks, 256, 0, stream>>>(src, dst, (int)n);
    return dst;
  };

  _Float16* conv1w16 = convert(conv1_w, 32 * 288);
  _Float16* capsw16  = convert(caps_w, 256 * 160);
  _Float16* fc1w16   = convert(fc1_w, (size_t)512 * 32);
  _Float16* fc2w16   = convert(fc2_w, (size_t)1024 * 512);
  _Float16* fc3w16   = convert(fc3_w, (size_t)21480 * 1024);

  const int bCin[3][4] = {{10, 42, 74, 106}, {41, 73, 105, 137}, {50, 82, 114, 146}};
  const int bCtot[3] = {138, 169, 178};
  const int bTout[3] = {41, 50, 53};
  const int bLen[3] = {2148, 428, 84};
  const int bLenOut[3] = {428, 84, 16};
  _Float16* dw16[3][4];
  _Float16* tw16[3];
  for (int blk = 0; blk < 3; ++blk) {
    for (int i = 0; i < 4; ++i)
      dw16[blk][i] = convert((const float*)d_in[denseBase + blk * 19 + i * 4 + 2],
                             (size_t)32 * bCin[blk][i] * 3);
    tw16[blk] = convert((const float*)d_in[denseBase + blk * 19 + 18],
                        (size_t)bTout[blk] * bCtot[blk]);
  }

  // 1) embedding + pooling
  embed_kernel<<<(BATCH * SEQL) / 8, 256, 0, stream>>>(x, emb_w, emb_b, h, pooled);
  // 2) attention + x_att output + h_att (f16, channel-major)
  att_kernel<<<(BATCH * SEQL + 255) / 256, 256, 0, stream>>>(h, pooled, conv2_w,
                                                             conv2_b, xatt, hatt);
  // 3) conv1 (WMMA implicit GEMM)
  conv1_wmma_kernel<<<dim3(45, 2, BATCH), 32, 0, stream>>>(conv1w16, hatt, conv1_b, h1);
  // 4) caps conv + squash (WMMA)
  caps_wmma_kernel<<<dim3(23, 16, BATCH), 32, 0, stream>>>(capsw16, h1, caps_b, prim);
  // 5) routing
  zero_f32_kernel<<<8, 256, 0, stream>>>(usum, 64 * 32);
  zero_f32_kernel<<<8, 256, 0, stream>>>(sbuf, 64 * 32);
  routing_pass1<<<dim3(45, BATCH), 256, 0, stream>>>(prim, rW, usum);
  routing_pass2<<<dim3(45, BATCH), 256, 0, stream>>>(prim, rW, usum, rb, sbuf);
  routing_final<<<BATCH, 32, 0, stream>>>(sbuf, x1f, x1h);
  // 6) reconstruction MLP (WMMA)
  gemm_wmma_kernel<<<dim3(32, 4), 32, 0, stream>>>(x1h, 32, fc1w16, 32, fc1_b,
                                                   g1f, g1h, 64, 512, 32, 1);
  gemm_wmma_kernel<<<dim3(64, 4), 32, 0, stream>>>(g1h, 512, fc2w16, 512, fc2_b,
                                                   g2f, g2h, 64, 1024, 512, 1);
  gemm_wmma_kernel<<<dim3(1343, 4), 32, 0, stream>>>(g2h, 1024, fc3w16, 1024, fc3_b,
                                                     xgen, (_Float16*)nullptr,
                                                     64, 21480, 1024, 2);
  // 7) out1 head
  head1_kernel<<<BATCH, 32, 0, stream>>>(x1f, lin1a_w, lin1a_b, lin1b_w, lin1b_b, out1);
  // 8) DenseNet branch on x_gen
  feat_init_kernel<<<(BATCH * DIN * SEQL + 255) / 256, 256, 0, stream>>>(xgen, featA);
  for (int blk = 0; blk < 3; ++blk) {
    int L = bLen[blk], Ctot = bCtot[blk], tout = bTout[blk];
    for (int i = 0; i < 4; ++i) {
      int cin = bCin[blk][i];
      const float* g  = (const float*)d_in[denseBase + blk * 19 + i * 4 + 0];
      const float* be = (const float*)d_in[denseBase + blk * 19 + i * 4 + 1];
      const float* bi = (const float*)d_in[denseBase + blk * 19 + i * 4 + 3];
      dense_conv_wmma_kernel<<<dim3((L + 15) / 16, 2, BATCH), 32, 0, stream>>>(
          dw16[blk][i], featA, g, be, bi, featA,
          cin, Ctot, Ctot, cin, 32, L, 3, 1, cin * 3);
    }
    const float* tg = (const float*)d_in[denseBase + blk * 19 + 16];
    const float* tb = (const float*)d_in[denseBase + blk * 19 + 17];
    dense_conv_wmma_kernel<<<dim3((L + 15) / 16, (tout + 15) / 16, BATCH), 32, 0,
                             stream>>>(tw16[blk], featA, tg, tb, nullptr, featT,
                                       Ctot, Ctot, tout, 0, tout, L, 1, 0, Ctot);
    int nextCtot = (blk < 2) ? bCtot[blk + 1] : 53;
    int Lo = bLenOut[blk];
    avgpool_kernel<<<(BATCH * tout * Lo + 255) / 256, 256, 0, stream>>>(
        featT, featA, tout, L, Lo, nextCtot);
  }
  // 9) x2 head
  head2_kernel<<<BATCH, 64, 0, stream>>>(featA, lin2a_w, lin2a_b, lin2b_w, lin2b_b, x2);
}
